// Transformer_82214263980391
// MI455X (gfx1250) — matrix operand order repaired and checked by reading, not compiled, not run on hardware
//
#include <hip/hip_runtime.h>
#include <hip/hip_bf16.h>

// ---- model dims (match reference) -----------------------------------------
#define LAYERS 8
#define DMODEL 1024
#define VOCAB  32000
#define NHEAD  16
#define DHEAD  64
#define BATCH  2
#define SEQ    2048
#define DFF    4096
#define NTOK   (BATCH * SEQ)          // 4096 rows

typedef __attribute__((ext_vector_type(16))) _Float16 v16h;
typedef __attribute__((ext_vector_type(8)))  float    v8f;

#define WMMA_F32_F16(a, b, c) \
  __builtin_amdgcn_wmma_f32_16x16x32_f16(false, (a), false, (b), (short)0, (c), false, false)

// A-fragment K index for V_WMMA_F32_16X16X32_F16 (ISA 7.12.2, 16-bit A 16x32)
__device__ __forceinline__ int akidx(int v, int half) {
  return ((v >> 2) << 4) + half * 8 + ((v & 3) << 1);
}
// B-fragment K index (same K order as the A fragment: K = 16*(v/4) + 8*half + 2*(v%4))
__device__ __forceinline__ int bkidx(int v, int half) {
  return ((v >> 2) << 4) + half * 8 + ((v & 3) << 1);
}

// ---- DPP16 cross-lane reductions over 16 lanes (pure VALU, no LDS) --------
// quad_perm(1,0,3,2)=0xB1 (xor1), quad_perm(2,3,0,1)=0x4E (xor2),
// row_ror:4=0x124, row_ror:8=0x128 (ror8 == xor8 on a 16-lane ring).
template <int CTRL>
__device__ __forceinline__ float dpp_mov(float v) {
  return __int_as_float(
      __builtin_amdgcn_update_dpp(0, __float_as_int(v), CTRL, 0xf, 0xf, true));
}
__device__ __forceinline__ float red16_max(float v) {
  v = fmaxf(v, dpp_mov<0x124>(v));
  v = fmaxf(v, dpp_mov<0x128>(v));
  v = fmaxf(v, dpp_mov<0xB1>(v));
  v = fmaxf(v, dpp_mov<0x4E>(v));
  return v;
}
__device__ __forceinline__ float red16_sum(float v) {
  v += dpp_mov<0x124>(v);
  v += dpp_mov<0x128>(v);
  v += dpp_mov<0xB1>(v);
  v += dpp_mov<0x4E>(v);
  return v;
}

// ---- async global -> LDS staging (ASYNCcnt path), with host fallback ------
// GLOBAL_LOAD_ASYNC_TO_LDS_B128, GV mode: vdst = LDS byte-offset VGPR,
// vaddr = 64-bit global address, SADDR=off. LDS offset = low 32 bits of the
// generic LDS pointer (ISA flat->LDS aperture uses addr[31:0] directly).
#if defined(__AMDGCN__)
#define GEMM_ASYNC 1
#else
#define GEMM_ASYNC 0
#endif

__device__ __forceinline__ void copy16_g2l(const float* g, float* l) {
#if GEMM_ASYNC
  unsigned lds_off = (unsigned)(unsigned long long)(const void*)l;
  asm volatile("global_load_async_to_lds_b128 %0, %1, off"
               :: "v"(lds_off), "v"(g) : "memory");
#else
  *(float4*)l = *(const float4*)g;
#endif
}
__device__ __forceinline__ void wait_async_le8() {
#if GEMM_ASYNC
  asm volatile("s_wait_asynccnt 0x8" ::: "memory");
#endif
}
__device__ __forceinline__ void wait_async_0() {
#if GEMM_ASYNC
  asm volatile("s_wait_asynccnt 0x0" ::: "memory");
#endif
}

// ---------------------------------------------------------------------------
// Embedding: x[b,s,:] = emb[tokens[b,s]] + pos[s]
// ---------------------------------------------------------------------------
__global__ __launch_bounds__(256) void embed_kernel(
    const int* __restrict__ tokens, const float* __restrict__ emb,
    const float* __restrict__ pos, float* __restrict__ X) {
  size_t i = (size_t)blockIdx.x * 256 + threadIdx.x;   // over NTOK*DMODEL
  int d = (int)(i % DMODEL);
  size_t tokidx = i / DMODEL;                          // 0..4095
  int s = (int)(tokidx % SEQ);
  int tk = tokens[tokidx];
  X[i] = emb[(size_t)tk * DMODEL + d] + pos[(size_t)s * DMODEL + d];
}

// ---------------------------------------------------------------------------
// LayerNorm (unbiased variance, ddof=1) — one block per token row
// ---------------------------------------------------------------------------
__global__ __launch_bounds__(256) void layernorm_kernel(
    const float* __restrict__ X, const float* __restrict__ gain,
    const float* __restrict__ bias, float* __restrict__ Y) {
  __shared__ float red[256];
  const int tid = threadIdx.x;
  const size_t row = blockIdx.x;
  const float* x = X + row * DMODEL;

  float vals[DMODEL / 256];
  float s = 0.f, s2 = 0.f;
#pragma unroll
  for (int i = 0; i < DMODEL / 256; ++i) {
    float v = x[tid + i * 256];
    vals[i] = v;
    s += v;
    s2 += v * v;
  }
  red[tid] = s;
  __syncthreads();
  for (int st = 128; st > 0; st >>= 1) {
    if (tid < st) red[tid] += red[tid + st];
    __syncthreads();
  }
  float mean = red[0] / (float)DMODEL;
  __syncthreads();
  red[tid] = s2;
  __syncthreads();
  for (int st = 128; st > 0; st >>= 1) {
    if (tid < st) red[tid] += red[tid + st];
    __syncthreads();
  }
  float var = (red[0] - (float)DMODEL * mean * mean) / (float)(DMODEL - 1);
  float inv = rsqrtf(var + 1e-9f);
#pragma unroll
  for (int i = 0; i < DMODEL / 256; ++i) {
    int col = tid + i * 256;
    Y[row * DMODEL + col] = (vals[i] - mean) * inv * gain[col] + bias[col];
  }
}

// ---------------------------------------------------------------------------
// Tiled WMMA GEMM: C[M,N] = act(A[M,K] @ W[K,N] + bias[N]) (+ residual)
// Block tile 128x128, K-tile 32. Raw f32 tiles staged to LDS via async
// b128 copies (double-buffered, ASYNCcnt); f32->f16 conversion happens at
// fragment read and co-executes with v_wmma. Padded LDS strides (36 / 132
// f32) keep fragment reads bank-conflict free and b128 chunks 16B-aligned.
// ---------------------------------------------------------------------------
#define AS_STRIDE 36
#define BS_STRIDE 132

template <bool SILU, bool RES>
__global__ __launch_bounds__(256) void gemm_wmma(
    const float* __restrict__ A, const float* __restrict__ W,
    const float* __restrict__ bias, const float* __restrict__ R,
    float* __restrict__ C, int M, int N, int K) {
  __shared__ float AsBuf[2][128 * AS_STRIDE];
  __shared__ float BsBuf[2][32 * BS_STRIDE];

  const int tid  = threadIdx.x;
  const int lane = tid & 31;
  const int wave = tid >> 5;
  const int wm   = wave & 1;    // 0..1 : 64-row slab
  const int wn   = wave >> 1;   // 0..3 : 32-col slab
  const int half = lane >> 4;
  const int lr   = lane & 15;
  const int m0   = blockIdx.y * 128;
  const int n0   = blockIdx.x * 128;

  v8f acc[4][2];
#pragma unroll
  for (int mf = 0; mf < 4; ++mf)
#pragma unroll
    for (int nf = 0; nf < 2; ++nf)
#pragma unroll
      for (int r = 0; r < 8; ++r) acc[mf][nf][r] = 0.f;

  // stage one 128x32 A-tile + 32x128 W-tile (b128 chunks, 8 per thread)
  auto stage = [&](int kt, int buf) {
#pragma unroll
    for (int t = 0; t < 4; ++t) {
      int ci = tid + t * 256;                  // 0..1023 chunks
      int r = ci >> 3, c4 = (ci & 7) << 2;
      copy16_g2l(&A[(size_t)(m0 + r) * K + kt + c4], &AsBuf[buf][r * AS_STRIDE + c4]);
    }
#pragma unroll
    for (int t = 0; t < 4; ++t) {
      int ci = tid + t * 256;
      int r = ci >> 5, c4 = (ci & 31) << 2;
      copy16_g2l(&W[(size_t)(kt + r) * N + n0 + c4], &BsBuf[buf][r * BS_STRIDE + c4]);
    }
  };

  stage(0, 0);
  int buf = 0;
  for (int kt = 0; kt < K; kt += 32, buf ^= 1) {
    if (kt + 32 < K) {
      stage(kt + 32, buf ^ 1);
      wait_async_le8();   // in-order: first 8 (this tile) complete
    } else {
      wait_async_0();
    }
    __syncthreads();

    const float* As32 = AsBuf[buf];
    const float* Bs32 = BsBuf[buf];

    v16h afr[4];
#pragma unroll
    for (int mf = 0; mf < 4; ++mf) {
      int row = wm * 64 + mf * 16 + lr;
      float4 a0 = *(const float4*)&As32[row * AS_STRIDE + half * 8];
      float4 a1 = *(const float4*)&As32[row * AS_STRIDE + half * 8 + 4];
      float4 a2 = *(const float4*)&As32[row * AS_STRIDE + 16 + half * 8];
      float4 a3 = *(const float4*)&As32[row * AS_STRIDE + 16 + half * 8 + 4];
      afr[mf][0] = (_Float16)a0.x;  afr[mf][1] = (_Float16)a0.y;
      afr[mf][2] = (_Float16)a0.z;  afr[mf][3] = (_Float16)a0.w;
      afr[mf][4] = (_Float16)a1.x;  afr[mf][5] = (_Float16)a1.y;
      afr[mf][6] = (_Float16)a1.z;  afr[mf][7] = (_Float16)a1.w;
      afr[mf][8] = (_Float16)a2.x;  afr[mf][9] = (_Float16)a2.y;
      afr[mf][10] = (_Float16)a2.z; afr[mf][11] = (_Float16)a2.w;
      afr[mf][12] = (_Float16)a3.x; afr[mf][13] = (_Float16)a3.y;
      afr[mf][14] = (_Float16)a3.z; afr[mf][15] = (_Float16)a3.w;
    }
    v16h bfr[2];
#pragma unroll
    for (int nf = 0; nf < 2; ++nf) {
      int col = wn * 32 + nf * 16 + lr;
#pragma unroll
      for (int v = 0; v < 8; ++v) {
        int k = bkidx(v, half);
        bfr[nf][2 * v]     = (_Float16)Bs32[k * BS_STRIDE + col];
        bfr[nf][2 * v + 1] = (_Float16)Bs32[(k + 1) * BS_STRIDE + col];
      }
    }
#pragma unroll
    for (int mf = 0; mf < 4; ++mf)
#pragma unroll
      for (int nf = 0; nf < 2; ++nf)
        acc[mf][nf] = WMMA_F32_F16(afr[mf], bfr[nf], acc[mf][nf]);
    __syncthreads();
  }

  // epilogue: bias (+SiLU) (+residual)
#pragma unroll
  for (int mf = 0; mf < 4; ++mf)
#pragma unroll
    for (int nf = 0; nf < 2; ++nf)
#pragma unroll
      for (int r = 0; r < 8; ++r) {
        int gm = m0 + wm * 64 + mf * 16 + r + half * 8;
        int gn = n0 + wn * 32 + nf * 16 + lr;
        float val = acc[mf][nf][r] + bias[gn];
        if (SILU) val = val / (1.f + __expf(-val));
        if (RES) val += R[(size_t)gm * N + gn];
        C[(size_t)gm * N + gn] = val;
      }
}

// ---------------------------------------------------------------------------
// Flash attention: one wave per (b, h, 16-query tile). Causal, online softmax
// with DPP16 reductions (no LDS in softmax); P staged through per-wave LDS
// (DS ops are in-order per wave); logits + P@V via v_wmma.
// ---------------------------------------------------------------------------
__global__ __launch_bounds__(256) void attn_flash(
    const float* __restrict__ Q, const float* __restrict__ Kb,
    const float* __restrict__ Vb, float* __restrict__ O) {
  __shared__ _Float16 Pl[8][16 * 32];     // per-wave P staging

  const int lane = threadIdx.x & 31;
  const int wv   = threadIdx.x >> 5;
  const int half = lane >> 4;
  const int lr   = lane & 15;

  const int QT  = SEQ / 16;               // 128 query tiles per (b,h)
  int wid = blockIdx.x * 8 + wv;          // 0..4095
  int qt  = wid % QT;
  int h   = (wid / QT) % NHEAD;
  int b   = wid / (QT * NHEAD);
  const int qbase = qt * 16;

  const size_t rowbase = (size_t)b * SEQ;

  // Q fragments (two k=32 chunks of DH=64), softmax scale folded in
  v16h qf[2];
  {
    const float* qr = Q + (rowbase + qbase + lr) * DMODEL + h * DHEAD;
#pragma unroll
    for (int c = 0; c < 2; ++c)
#pragma unroll
      for (int v = 0; v < 8; ++v) {
        int k = c * 32 + akidx(v, half);
        qf[c][2 * v]     = (_Float16)(qr[k] * 0.125f);
        qf[c][2 * v + 1] = (_Float16)(qr[k + 1] * 0.125f);
      }
  }

  v8f oacc[4];
#pragma unroll
  for (int g = 0; g < 4; ++g)
#pragma unroll
    for (int r = 0; r < 8; ++r) oacc[g][r] = 0.f;
  float mrow[8], lrow[8];
#pragma unroll
  for (int r = 0; r < 8; ++r) { mrow[r] = -1e30f; lrow[r] = 0.f; }

  const int jbmax = (qbase + 15) / 32;
  for (int jb = 0; jb <= jbmax; ++jb) {
    const int kbb = jb * 32;

    // logits 16x32 = Q(16x64) @ K^T(64x32), two 16x16 C frags
    v8f lg[2];
#pragma unroll
    for (int ng = 0; ng < 2; ++ng) {
      const float* kr = Kb + (rowbase + kbb + ng * 16 + lr) * DMODEL + h * DHEAD;
      v16h bf0, bf1;
#pragma unroll
      for (int v = 0; v < 8; ++v) {
        int k = bkidx(v, half);
        bf0[2 * v]     = (_Float16)kr[k];
        bf0[2 * v + 1] = (_Float16)kr[k + 1];
        bf1[2 * v]     = (_Float16)kr[32 + k];
        bf1[2 * v + 1] = (_Float16)kr[32 + k + 1];
      }
      v8f z;
#pragma unroll
      for (int r = 0; r < 8; ++r) z[r] = 0.f;
      lg[ng] = WMMA_F32_F16(qf[0], bf0, z);
      lg[ng] = WMMA_F32_F16(qf[1], bf1, lg[ng]);
    }

    // causal mask + online softmax (rows live across 16 lanes of a half-wave)
    float pv[2][8];
#pragma unroll
    for (int r = 0; r < 8; ++r) {
      int qi = qbase + r + half * 8;
      float c0 = lg[0][r];
      float c1 = lg[1][r];
      if (kbb + lr > qi) c0 = -1e30f;
      if (kbb + 16 + lr > qi) c1 = -1e30f;
      float vmax = red16_max(fmaxf(c0, c1));
      float nm = fmaxf(mrow[r], vmax);
      float sc = __expf(mrow[r] - nm);
      float p0 = __expf(c0 - nm);
      float p1 = __expf(c1 - nm);
      float ps = red16_sum(p0 + p1);
      lrow[r] = lrow[r] * sc + ps;
      mrow[r] = nm;
#pragma unroll
      for (int g = 0; g < 4; ++g) oacc[g][r] *= sc;
      pv[0][r] = p0;
      pv[1][r] = p1;
    }

    // C-layout P -> LDS row-major [16][32] (DS ops are in-order per wave)
#pragma unroll
    for (int r = 0; r < 8; ++r) {
      int m = r + half * 8;
      Pl[wv][m * 32 + lr]      = (_Float16)pv[0][r];
      Pl[wv][m * 32 + 16 + lr] = (_Float16)pv[1][r];
    }
#if defined(__AMDGCN__)
    asm volatile("s_wait_dscnt 0" ::: "memory");
#endif

    // P as A-fragment (16x32)
    v16h pf;
#pragma unroll
    for (int v = 0; v < 8; ++v) {
      int k = akidx(v, half);
      pf[2 * v]     = Pl[wv][lr * 32 + k];
      pf[2 * v + 1] = Pl[wv][lr * 32 + k + 1];
    }

    // pre(16x64) += P(16x32) @ V(32x64): four dh groups of 16
#pragma unroll
    for (int g = 0; g < 4; ++g) {
      v16h vf;
#pragma unroll
      for (int v = 0; v < 8; ++v) {
        int k = bkidx(v, half);   // key index within block
        vf[2 * v]     = (_Float16)Vb[(rowbase + kbb + k) * DMODEL + h * DHEAD + g * 16 + lr];
        vf[2 * v + 1] = (_Float16)Vb[(rowbase + kbb + k + 1) * DMODEL + h * DHEAD + g * 16 + lr];
      }
      oacc[g] = WMMA_F32_F16(pf, vf, oacc[g]);
    }
  }

#pragma unroll
  for (int g = 0; g < 4; ++g)
#pragma unroll
    for (int r = 0; r < 8; ++r) {
      int qi = qbase + r + half * 8;
      O[(rowbase + qi) * DMODEL + h * DHEAD + g * 16 + lr] = oacc[g][r] / lrow[r];
    }
}

// ---------------------------------------------------------------------------
extern "C" void kernel_launch(void* const* d_in, const int* in_sizes, int n_in,
                              void* d_out, int out_size, void* d_ws, size_t ws_size,
                              hipStream_t stream) {
  (void)in_sizes; (void)n_in; (void)out_size; (void)ws_size;
  const int*   tokens = (const int*)d_in[0];
  const float* emb    = (const float*)d_in[1];
  const float* pos    = (const float*)d_in[2];
  const float* Wq     = (const float*)d_in[3];
  const float* bq     = (const float*)d_in[4];
  const float* Wk     = (const float*)d_in[5];
  const float* bk     = (const float*)d_in[6];
  const float* Wv     = (const float*)d_in[7];
  const float* bv     = (const float*)d_in[8];
  const float* Wo     = (const float*)d_in[9];
  const float* bo     = (const float*)d_in[10];
  const float* W_up   = (const float*)d_in[11];
  const float* b_up   = (const float*)d_in[12];
  const float* W_down = (const float*)d_in[13];
  const float* b_down = (const float*)d_in[14];
  const float* ln1_g  = (const float*)d_in[15];
  const float* ln1_b  = (const float*)d_in[16];
  const float* ln2_g  = (const float*)d_in[17];
  const float* ln2_b  = (const float*)d_in[18];
  const float* W_un   = (const float*)d_in[19];
  const float* b_un   = (const float*)d_in[20];
  float* out = (float*)d_out;

  const size_t TD = (size_t)NTOK * DMODEL;   // 4M floats
  float* ws  = (float*)d_ws;
  float* x   = ws;
  float* hb  = ws + 1 * TD;
  float* q   = ws + 2 * TD;
  float* k   = ws + 3 * TD;
  float* v   = ws + 4 * TD;
  float* pre = ws + 5 * TD;
  float* u   = ws + 6 * TD;                  // NTOK x DFF

  // x = emb[tokens] + pos
  embed_kernel<<<(TD / 256), 256, 0, stream>>>(tokens, emb, pos, x);

  const dim3 gD(DMODEL / 128, NTOK / 128);   // N=1024 GEMMs
  const dim3 gF(DFF / 128, NTOK / 128);      // N=4096 GEMM
  const dim3 gV(VOCAB / 128, NTOK / 128);    // N=32000 GEMM
  const int attnBlocks = (BATCH * NHEAD * (SEQ / 16)) / 8;   // 512

  for (int i = 0; i < LAYERS; ++i) {
    const size_t wOff  = (size_t)i * DMODEL * DMODEL;
    const size_t fOff  = (size_t)i * DMODEL * DFF;
    const size_t dOff  = (size_t)i * DMODEL;
    const size_t fbOff = (size_t)i * DFF;

    // h = LN1(x)
    layernorm_kernel<<<NTOK, 256, 0, stream>>>(x, ln1_g + dOff, ln1_b + dOff, hb);
    // q,k,v projections
    gemm_wmma<false, false><<<gD, 256, 0, stream>>>(hb, Wq + wOff, bq + dOff, nullptr, q, NTOK, DMODEL, DMODEL);
    gemm_wmma<false, false><<<gD, 256, 0, stream>>>(hb, Wk + wOff, bk + dOff, nullptr, k, NTOK, DMODEL, DMODEL);
    gemm_wmma<false, false><<<gD, 256, 0, stream>>>(hb, Wv + wOff, bv + dOff, nullptr, v, NTOK, DMODEL, DMODEL);
    // pre = softmax(q k^T / sqrt(dh), causal) v
    attn_flash<<<attnBlocks, 256, 0, stream>>>(q, k, v, pre);
    // x = x + pre @ Wo + bo
    gemm_wmma<false, true><<<gD, 256, 0, stream>>>(pre, Wo + wOff, bo + dOff, x, x, NTOK, DMODEL, DMODEL);
    // h = LN2(x)
    layernorm_kernel<<<NTOK, 256, 0, stream>>>(x, ln2_g + dOff, ln2_b + dOff, hb);
    // u = silu(h @ W_up + b_up)
    gemm_wmma<true, false><<<gF, 256, 0, stream>>>(hb, W_up + fOff, b_up + fbOff, nullptr, u, NTOK, DFF, DMODEL);
    // x = x + u @ W_down + b_down
    gemm_wmma<false, true><<<gD, 256, 0, stream>>>(u, W_down + fOff, b_down + dOff, x, x, NTOK, DMODEL, DFF);
  }

  // logits = x @ W_un + b_un
  gemm_wmma<false, false><<<gV, 256, 0, stream>>>(x, W_un, b_un, nullptr, out, NTOK, VOCAB, DMODEL);
}
